// GRU_24129126269612
// MI455X (gfx1250) — matrix operand order repaired and checked by reading, not compiled, not run on hardware
//
#include <hip/hip_runtime.h>
#include <stdint.h>

// ---------------- problem constants ----------------
#define L_SEQ   2048
#define BATCH   64
#define D_INP   256
#define LATENT  512
#define CATDIM  768          // D_INP + LATENT
#define NWG     16           // persistent workgroups (1 per WGP)
#define NTHR    256          // 8 waves of 32
#define R_SL    48           // CATDIM / NWG   (per-WG slice of r rows)
#define H_SL    32           // LATENT / NWG   (per-WG slice of latent rows)

// ---------------- LDS layout (bytes) ----------------
// rows padded to 776 bf16 elements = 1552B to break bank alignment
#define ROWB    1552u
#define CATOFF  0u
#define WROFF   (CATOFF + 64u * ROWB)     //  99328
#define WZOFF   (WROFF + 48u * ROWB)      // 173824
#define WHOFF   (WZOFF + 32u * ROWB)      // 223488
#define ZOFF    (WHOFF + 32u * ROWB)      // 273152
#define SMEMSZ  (ZOFF + 64u * 32u * 4u)   // 281344 < 320KB per WGP

// ---------------- workspace layout (bytes) ----------------
#define WS_CNT  0
#define WS_H    256                                   // bf16 h   [64][512]
#define WS_R    (256 + BATCH * LATENT * 2)            // bf16 r   [64][768]
#define WS_WZ   (WS_R + BATCH * CATDIM * 2)           // bf16 Wz  [512][768]
#define WS_WR   (WS_WZ + LATENT * CATDIM * 2)         // bf16 Wr  [768][768]
#define WS_WH   (WS_WR + CATDIM * CATDIM * 2)         // bf16 Wh  [512][768]

typedef __attribute__((ext_vector_type(16))) __bf16 v16bf;
typedef __attribute__((ext_vector_type(8)))  float  v8f;

// ---------------- bf16 helpers ----------------
__device__ __forceinline__ unsigned short bfbits(float f) {
    __bf16 b = (__bf16)f;
    return __builtin_bit_cast(unsigned short, b);
}
__device__ __forceinline__ unsigned pack2(float a, float b) {
    return (unsigned)bfbits(a) | ((unsigned)bfbits(b) << 16);
}
// packed bf16 multiply: one V_PK_MUL_BF16 instead of unpack/mul/repack
__device__ __forceinline__ unsigned mul2(unsigned a, unsigned b) {
    unsigned o;
    asm("v_pk_mul_bf16 %0, %1, %2" : "=v"(o) : "v"(a), "v"(b));
    return o;
}

// ---------------- WMMA fragment loads from LDS ----------------
// A (16-bit, 16x32): lane half h holds contiguous K run [16h, 16h+15]
//   -> two b128 loads, 16B apart.
__device__ __forceinline__ v16bf ld_frag_a(const uint8_t* smem, unsigned off) {
    union { uint4 u[2]; v16bf v; } r;
    r.u[0] = *(const uint4*)(smem + off);
    r.u[1] = *(const uint4*)(smem + off + 16);
    return r.v;
}
// B (16-bit, 32x16): lane half h holds contiguous K run [16h, 16h+15]
//   -> 32 contiguous bytes.
__device__ __forceinline__ v16bf ld_frag_b(const uint8_t* smem, unsigned off) {
    union { uint4 u[2]; v16bf v; } r;
    r.u[0] = *(const uint4*)(smem + off);
    r.u[1] = *(const uint4*)(smem + off + 16);
    return r.v;
}

// ---------------- device-scope barrier across NWG persistent WGs ----------------
__device__ __forceinline__ void grid_barrier(unsigned* cnt, unsigned target) {
    __threadfence();            // publish prior global writes (device scope)
    __syncthreads();
    if (threadIdx.x == 0) {
        __hip_atomic_fetch_add(cnt, 1u, __ATOMIC_RELEASE, __HIP_MEMORY_SCOPE_AGENT);
        while (__hip_atomic_load(cnt, __ATOMIC_ACQUIRE, __HIP_MEMORY_SCOPE_AGENT) < target) {
            __builtin_amdgcn_s_sleep(2);
        }
    }
    __syncthreads();
    __builtin_amdgcn_fence(__ATOMIC_ACQUIRE, "agent");  // invalidate stale near caches
}

// ---------------- fp32 -> bf16 weight prep ----------------
__global__ void cvt_f32_bf16(const float* __restrict__ s, __bf16* __restrict__ d, int n4) {
    int i = blockIdx.x * blockDim.x + threadIdx.x;
    if (i < n4) {
        float4 v = ((const float4*)s)[i];
        uint2 p;
        p.x = pack2(v.x, v.y);
        p.y = pack2(v.z, v.w);
        ((uint2*)d)[i] = p;
    }
}

// ---------------- persistent GRU kernel ----------------
__global__ __launch_bounds__(NTHR) void gru_persistent(
    const float* __restrict__ x, float* __restrict__ out,
    const __bf16* __restrict__ wz, const __bf16* __restrict__ wr,
    const __bf16* __restrict__ wh,
    __bf16* __restrict__ hbuf,     // [64][512] bf16, L2-resident h state
    __bf16* __restrict__ rbuf,     // [64][768] bf16, r gate exchange
    unsigned* __restrict__ cnt)
{
    extern __shared__ uint8_t smem[];
    const int g    = blockIdx.x;          // WG slice id 0..15
    const int tid  = threadIdx.x;
    const int wave = tid >> 5;
    const int lane = tid & 31;
    const int half = lane >> 4;           // K-half of the wave
    const int nrow = lane & 15;           // row-in-tile (A) / col-in-tile (B,C,D)
    const int mt   = wave & 3;            // fixed batch tile per wave (64 = 4x16)
    const int ntb  = wave >> 2;           // 0 or 1

    // ---- one-time: stage this WG's weight slices into LDS (bf16, padded rows) ----
    {
        const uint4* s = (const uint4*)(wr + (size_t)g * R_SL * CATDIM);
        for (int i = tid; i < 48 * 96; i += NTHR) {
            int r = i / 96, c = i % 96;
            *(uint4*)(smem + WROFF + (unsigned)r * ROWB + (unsigned)c * 16u) = s[r * 96 + c];
        }
        s = (const uint4*)(wz + (size_t)g * H_SL * CATDIM);
        for (int i = tid; i < 32 * 96; i += NTHR) {
            int r = i / 96, c = i % 96;
            *(uint4*)(smem + WZOFF + (unsigned)r * ROWB + (unsigned)c * 16u) = s[r * 96 + c];
        }
        s = (const uint4*)(wh + (size_t)g * H_SL * CATDIM);
        for (int i = tid; i < 32 * 96; i += NTHR) {
            int r = i / 96, c = i % 96;
            *(uint4*)(smem + WHOFF + (unsigned)r * ROWB + (unsigned)c * 16u) = s[r * 96 + c];
        }
    }
    __syncthreads();

    // per-wave constant fragment base offsets
    const unsigned abase = CATOFF + (unsigned)(mt * 16 + nrow) * ROWB + 32u * (unsigned)half;
    const unsigned bsel  = 32u * (unsigned)half;

    // activation emit helpers
    auto emit_r = [&](const v8f& acc, int nt) {
        #pragma unroll
        for (int v = 0; v < 8; ++v) {
            float sg = 1.0f / (1.0f + __expf(-acc[v]));
            int m = mt * 16 + v + 8 * half;               // C/D: M = v + 8*(lane/16)
            rbuf[m * CATDIM + g * R_SL + nt * 16 + nrow] = (__bf16)sg;
        }
    };
    auto emit_z = [&](const v8f& acc, int zt) {           // zt in {0,1}
        #pragma unroll
        for (int v = 0; v < 8; ++v) {
            float sg = 1.0f / (1.0f + __expf(-acc[v]));
            int m = mt * 16 + v + 8 * half;
            *(float*)(smem + ZOFF +
                      ((unsigned)m * 32u + (unsigned)zt * 16u + (unsigned)nrow) * 4u) = sg;
        }
    };

    unsigned epoch = 0;
    for (int t = 0; t < L_SEQ; ++t) {
        // ---------- phase 0: cat = [bf16(x_t) | h] into LDS ----------
        {
            const float4* xs = (const float4*)(x + (size_t)t * BATCH * D_INP);
            for (int i = tid; i < 64 * 64; i += NTHR) {      // 64 rows x 64 float4
                int r = i >> 6, c = i & 63;
                float4 v = xs[r * 64 + c];
                uint2 p;
                p.x = pack2(v.x, v.y);
                p.y = pack2(v.z, v.w);
                *(uint2*)(smem + CATOFF + (unsigned)r * ROWB + (unsigned)c * 8u) = p;
            }
            // h state (already bf16): async copy straight into LDS cat[:,256:768]
            for (int i = tid; i < 64 * 64; i += NTHR) {      // 64 rows x 64 x b128
                int r = i >> 6, c = i & 63;
                unsigned loff = CATOFF + (unsigned)r * ROWB + 512u + (unsigned)c * 16u;
                const uint8_t* gp = (const uint8_t*)hbuf + (size_t)r * 1024 + (size_t)c * 16;
                asm volatile("global_load_async_to_lds_b128 %0, %1, off"
                             :: "v"(loff), "v"(gp) : "memory");
            }
            // prefetch next timestep's x under the coming WMMA work
            if (t + 1 < L_SEQ) {
                const char* nx = (const char*)(x + (size_t)(t + 1) * BATCH * D_INP)
                                 + (size_t)tid * 256;
                __builtin_prefetch(nx, 0, 0);
                __builtin_prefetch(nx + 128, 0, 0);
            }
            asm volatile("s_wait_asynccnt 0x0" ::: "memory");
        }
        __syncthreads();

        // ---------- GEMM 1&2 fused: r_slice/z_slice = sigmoid(cat @ W^T) ----------
        // 5 N-tiles (3 r-tiles, 2 z-tiles) x 4 M-tiles = 20 tiles over 8 waves.
        // Wave-uniform split: waves with ntb==0 run 3 tiles, ntb==1 run 2 tiles,
        // as straight-line loops (no per-iteration exec masking around WMMA).
        if (ntb == 0) {
            v8f acc0 = {}, acc1 = {}, acc2 = {};
            const unsigned b0 = WROFF + (unsigned)(0 * 16 + nrow) * ROWB + bsel;  // r nt=0
            const unsigned b1 = WROFF + (unsigned)(2 * 16 + nrow) * ROWB + bsel;  // r nt=2
            const unsigned b2 = WZOFF + (unsigned)(1 * 16 + nrow) * ROWB + bsel;  // z zt=1
            for (int kb = 0; kb < 24; ++kb) {
                const unsigned kby = (unsigned)kb * 64u;
                v16bf a  = ld_frag_a(smem, abase + kby);
                v16bf w0 = ld_frag_b(smem, b0 + kby);
                v16bf w1 = ld_frag_b(smem, b1 + kby);
                v16bf w2 = ld_frag_b(smem, b2 + kby);
                acc0 = __builtin_amdgcn_wmma_f32_16x16x32_bf16(false, a, false, w0,
                                                               (short)0, acc0, false, false);
                acc1 = __builtin_amdgcn_wmma_f32_16x16x32_bf16(false, a, false, w1,
                                                               (short)0, acc1, false, false);
                acc2 = __builtin_amdgcn_wmma_f32_16x16x32_bf16(false, a, false, w2,
                                                               (short)0, acc2, false, false);
            }
            emit_r(acc0, 0);
            emit_r(acc1, 2);
            emit_z(acc2, 1);
        } else {
            v8f acc0 = {}, acc1 = {};
            const unsigned b0 = WROFF + (unsigned)(1 * 16 + nrow) * ROWB + bsel;  // r nt=1
            const unsigned b1 = WZOFF + (unsigned)(0 * 16 + nrow) * ROWB + bsel;  // z zt=0
            for (int kb = 0; kb < 24; ++kb) {
                const unsigned kby = (unsigned)kb * 64u;
                v16bf a  = ld_frag_a(smem, abase + kby);
                v16bf w0 = ld_frag_b(smem, b0 + kby);
                v16bf w1 = ld_frag_b(smem, b1 + kby);
                acc0 = __builtin_amdgcn_wmma_f32_16x16x32_bf16(false, a, false, w0,
                                                               (short)0, acc0, false, false);
                acc1 = __builtin_amdgcn_wmma_f32_16x16x32_bf16(false, a, false, w1,
                                                               (short)0, acc1, false, false);
            }
            emit_r(acc0, 1);
            emit_z(acc1, 0);
        }

        epoch += NWG;
        grid_barrier(cnt, epoch);   // all r slices visible device-wide

        // ---------- rc = r (full, from L2) * cat, in place in LDS ----------
        for (int i = tid; i < 64 * 96; i += NTHR) {          // 64 rows x 96 x 16B
            int r = i / 96, c = i % 96;
            uint8_t* lp = smem + CATOFF + (unsigned)r * ROWB + (unsigned)c * 16u;
            uint4 cu = *(uint4*)lp;
            uint4 ru = *(const uint4*)((const uint8_t*)rbuf + (size_t)r * 1536 + (size_t)c * 16);
            uint4 o;
            o.x = mul2(cu.x, ru.x);
            o.y = mul2(cu.y, ru.y);
            o.z = mul2(cu.z, ru.z);
            o.w = mul2(cu.w, ru.w);
            *(uint4*)lp = o;
        }
        __syncthreads();

        // ---------- GEMM 3: h_tilde_slice = tanh(rc @ Wh_slice^T); blend; store ----------
        {
            v8f acc = {};
            const int nl = ntb;                               // 2 N-tiles x 4 M-tiles = 8 waves
            const unsigned bbase = WHOFF + (unsigned)(nl * 16 + nrow) * ROWB + bsel;
            for (int kb = 0; kb < 24; ++kb) {
                const unsigned kby = (unsigned)kb * 64u;
                v16bf a = ld_frag_a(smem, abase + kby);
                v16bf b = ld_frag_b(smem, bbase + kby);
                acc = __builtin_amdgcn_wmma_f32_16x16x32_bf16(false, a, false, b,
                                                              (short)0, acc, false, false);
            }
            float* outT = out + (size_t)t * BATCH * LATENT;
            #pragma unroll
            for (int v = 0; v < 8; ++v) {
                int m   = mt * 16 + v + 8 * half;
                int cl  = nl * 16 + nrow;                     // col within slice [0,32)
                int col = g * H_SL + cl;                      // global latent col
                float xc = fminf(fmaxf(acc[v], -15.f), 15.f);
                float e  = __expf(-2.f * xc);
                float ht = (1.f - e) / (1.f + e);             // tanh
                float z  = *(const float*)(smem + ZOFF + ((unsigned)m * 32u + (unsigned)cl) * 4u);
                float ho = (float)hbuf[m * LATENT + col];
                float hn = ho + z * (ht - ho);                // (1-z)h + z*h~
                outT[m * LATENT + col] = hn;
                hbuf[m * LATENT + col] = (__bf16)hn;          // bf16 state for next step
            }
        }

        epoch += NWG;
        grid_barrier(cnt, epoch);   // h fully updated before next step's loads
    }
}

// ---------------- host launcher ----------------
extern "C" void kernel_launch(void* const* d_in, const int* in_sizes, int n_in,
                              void* d_out, int out_size, void* d_ws, size_t ws_size,
                              hipStream_t stream) {
    (void)in_sizes; (void)n_in; (void)out_size; (void)ws_size;
    const float* x  = (const float*)d_in[0];
    const float* Wz = (const float*)d_in[1];
    const float* Wr = (const float*)d_in[2];
    const float* Wh = (const float*)d_in[3];
    float* out = (float*)d_out;
    uint8_t* ws = (uint8_t*)d_ws;

    unsigned* cnt = (unsigned*)(ws + WS_CNT);
    __bf16* hbuf  = (__bf16*)(ws + WS_H);
    __bf16* rbuf  = (__bf16*)(ws + WS_R);
    __bf16* wzb   = (__bf16*)(ws + WS_WZ);
    __bf16* wrb   = (__bf16*)(ws + WS_WR);
    __bf16* whb   = (__bf16*)(ws + WS_WH);

    // zero the barrier counter + initial h state (h0 = 0), every call
    hipMemsetAsync(ws, 0, WS_R, stream);

    int n4;
    n4 = LATENT * CATDIM / 4;
    cvt_f32_bf16<<<(n4 + NTHR - 1) / NTHR, NTHR, 0, stream>>>(Wz, wzb, n4);
    n4 = CATDIM * CATDIM / 4;
    cvt_f32_bf16<<<(n4 + NTHR - 1) / NTHR, NTHR, 0, stream>>>(Wr, wrb, n4);
    n4 = LATENT * CATDIM / 4;
    cvt_f32_bf16<<<(n4 + NTHR - 1) / NTHR, NTHR, 0, stream>>>(Wh, whb, n4);

    hipFuncSetAttribute((const void*)gru_persistent,
                        hipFuncAttributeMaxDynamicSharedMemorySize, (int)SMEMSZ);
    gru_persistent<<<NWG, NTHR, SMEMSZ, stream>>>(x, out, wzb, wrb, whb, hbuf, rbuf, cnt);
}
